// VectorQuantizer_90563680403806
// MI455X (gfx1250) — matrix operand order repaired and checked by reading, not compiled, not run on hardware
//
#include <hip/hip_runtime.h>

typedef __attribute__((ext_vector_type(16))) _Float16 v16h;
typedef __attribute__((ext_vector_type(8)))  float    v8f;
typedef __attribute__((ext_vector_type(4)))  unsigned int v4u;
typedef __attribute__((ext_vector_type(8)))  int      v8i;
typedef __attribute__((ext_vector_type(4)))  int      v4i;

#define N_ROWS   65536     // 16*64*64
#define K_CODES  4096
#define C_DIM    32
#define HW       4096      // 64*64
#define CHW      131072    // 32*4096
#define KTILES   256       // 4096/16
#define BETA     0.25f

// workspace layout (bytes); total ~4.74 MB
#define PARTIAL_OFF 0          // 256 floats: per-block loss partials
#define ENORM_OFF   1024       // 4096 floats (16 KB)
#define BPACK_OFF   17408      // 131072 f16 (256 KB), WMMA-B lane layout
#define APACK_OFF   279552    // 2097152 f16 (4 MB), WMMA-A lane layout
#define IDX_OFF     4473856    // 65536 int32 argmin indices

// ---------------------------------------------------------------- ‖e_k‖²
__global__ void enorm_kernel(const float* __restrict__ table,
                             float* __restrict__ enorm) {
    int k = blockIdx.x * blockDim.x + threadIdx.x;
    if (k >= K_CODES) return;
    const float* r = table + k * C_DIM;
    float s = 0.f;
#pragma unroll
    for (int c = 0; c < C_DIM; ++c) s += r[c] * r[c];
    enorm[k] = s;
}

// ------------------------------------------------- pack table -> B layout
// B (32x16 f16, 8 VGPRs): lane l holds column n=l&15;
//   l<16 : v<4 -> K=2v,2v+1 ; v>=4 -> K=2v+8,2v+9    (K 0..7, 16..23)
//   l>=16: v<4 -> K=2v+8    ; v>=4 -> K=2v+16        (K 8..15, 24..31)
__global__ void packB_kernel(const float* __restrict__ table,
                             _Float16* __restrict__ bpack) {
    int tid  = blockIdx.x * blockDim.x + threadIdx.x;   // 65536 threads
    int tile = tid >> 8;            // 256 K-tiles
    int lane = (tid >> 3) & 31;
    int v    = tid & 7;
    int n    = tile * 16 + (lane & 15);
    int kb   = ((lane < 16) ? 0 : 8) + ((v < 4) ? (2 * v) : (2 * v + 8));
    _Float16* dst = bpack + tile * 512 + lane * 16 + 2 * v;
    dst[0] = (_Float16)table[n * C_DIM + kb];
    dst[1] = (_Float16)table[n * C_DIM + kb + 1];
}

// ---------------------------------------------------- pack z -> A layout
// A (16x32 f16, 8 VGPRs): lane m holds row m (m = lane&15);
//   l<16 : v<4 -> K=2v,2v+1 ; v>=4 -> K=2v+8,2v+9    (K 0..7, 16..23)
//   l>=16: v<4 -> K=2v+8    ; v>=4 -> K=2v+16        (K 8..15, 24..31)
__global__ void packA_kernel(const float* __restrict__ z,
                             _Float16* __restrict__ apack) {
    int tid  = blockIdx.x * blockDim.x + threadIdx.x;   // 1048576 threads
    int tile = tid >> 8;            // 4096 M-tiles
    int lane = (tid >> 3) & 31;
    int v    = tid & 7;
    int row  = tile * 16 + (lane & 15);
    int kb;
    if (lane < 16) kb = (v < 4) ? (2 * v)     : (2 * v + 8);
    else           kb = (v < 4) ? (2 * v + 8) : (2 * v + 16);
    int b  = row >> 12;             // row / 4096
    int hw = row & (HW - 1);
    long off = (long)b * CHW + (long)kb * HW + hw;       // z[b][kb][h][w]
    _Float16* dst = apack + tile * 512 + lane * 16 + 2 * v;
    dst[0] = (_Float16)z[off];
    dst[1] = (_Float16)z[off + HW]; // kb+1 -> +HW floats
}

// --------------------------- TDM descriptor helper (2-D tile, 8B units)
// D# per CDNA5 ISA; groups 2/3 zero (<=2D tensor).
__device__ __forceinline__ void tdm_load_2d(unsigned lds_byte_off,
                                            const void* gptr,
                                            unsigned dim0_u8,   // units of 8B
                                            unsigned dim1) {
    unsigned long long ga = (unsigned long long)gptr;
    v4u g0;
    g0.x = 1u;                                        // count=1, user desc
    g0.y = lds_byte_off;                              // lds_addr
    g0.z = (unsigned)(ga & 0xFFFFFFFFull);            // global_addr[31:0]
    g0.w = (unsigned)((ga >> 32) & 0x1FFFFFFull)      // global_addr[56:32]
         | (2u << 30);                                // type = 2 ("image")
    v8i g1;
    g1[0] = (int)(3u << 16);                          // wg_mask=0, data_size=8B
    g1[1] = (int)((dim0_u8 & 0xFFFFu) << 16);         // tensor_dim0[15:0]
    g1[2] = (int)(((dim0_u8 >> 16) & 0xFFFFu)         // tensor_dim0[31:16]
         | ((dim1 & 0xFFFFu) << 16));                 // tensor_dim1[15:0]
    g1[3] = (int)(((dim1 >> 16) & 0xFFFFu)            // tensor_dim1[31:16]
         | ((dim0_u8 & 0xFFFFu) << 16));              // tile_dim0
    g1[4] = (int)(dim1 & 0xFFFFu);                    // tile_dim1 (tile_dim2=0)
    g1[5] = (int)dim0_u8;                             // tensor_dim0_stride lo32
    g1[6] = 0;                                        // stride0 hi / stride1 lo
    g1[7] = 0;                                        // stride1 hi
    v4i zz4 = {0, 0, 0, 0};
    v8i zz8 = {0, 0, 0, 0, 0, 0, 0, 0};
    // 6-arg form (clang-23 / therock headers): (g0, g1, g2, g3, g4, cpol)
    __builtin_amdgcn_tensor_load_to_lds(g0, g1, zz4, zz4, zz8, 0);
}

// ---------------------------------- fused GEMM (WMMA) + argmin over 4096
// 256 threads = 8 waves; each wave owns TWO 16-row M-tiles (32 rows) so
// every B tile read from LDS feeds two independent WMMAs (fills the
// WMMA->VALU hazard slots and halves LDS traffic per row). Whole packed
// codebook (256 KB) + norms (16 KB) staged in LDS (272 KB of 320 KB/WGP)
// by the Tensor Data Mover (one wave issues, TENSORcnt tracks).
__global__ void __launch_bounds__(256)
vq_argmin_kernel(const _Float16* __restrict__ apack,
                 const _Float16* __restrict__ bpack,
                 const float* __restrict__ enorm,
                 int* __restrict__ out_idx) {
    extern __shared__ char smem[];
    _Float16* sB = (_Float16*)smem;                       // 262144 bytes
    float*    sE = (float*)(smem + 262144);               // 16384 bytes

    const int tid = threadIdx.x;

    // ---- TDM fill of LDS: codebook (256KB) + norms (16KB), wave 0 only
    if (tid < 32) {
        typedef __attribute__((address_space(3))) char as3char;
        unsigned lds0 = (unsigned)(unsigned long long)(as3char*)smem;
        tdm_load_2d(lds0,           bpack, 4096u, 8u);    // 4096*8*8B=256KB
        tdm_load_2d(lds0 + 262144u, enorm, 2048u, 1u);    // 2048*8B = 16KB
        __builtin_amdgcn_s_wait_tensorcnt(0);
    }
    __syncthreads();

    const int wave   = tid >> 5;
    const int lane   = tid & 31;
    const int tileA0 = (blockIdx.x * 8 + wave) * 2;       // first M-tile
    const int tileA1 = tileA0 + 1;                        // second M-tile

    const v16h a0 = *(const v16h*)(apack + tileA0 * 512 + lane * 16);
    const v16h a1 = *(const v16h*)(apack + tileA1 * 512 + lane * 16);

    float minv0[8], minv1[8];
    int   mini0[8], mini1[8];
#pragma unroll
    for (int j = 0; j < 8; ++j) {
        minv0[j] = 3.0e38f; mini0[j] = 0;
        minv1[j] = 3.0e38f; mini1[j] = 0;
    }

    const int nlane = lane & 15;   // codebook column this lane sees in C/D

#pragma unroll 2
    for (int t = 0; t < KTILES; ++t) {
        const v16h b = *(const v16h*)(sB + t * 512 + lane * 16);
        v8f c0 = {};
        v8f c1 = {};
        c0 = __builtin_amdgcn_wmma_f32_16x16x32_f16(
                 false, a0, false, b, (short)0, c0, false, false);
        c1 = __builtin_amdgcn_wmma_f32_16x16x32_f16(
                 false, a1, false, b, (short)0, c1, false, false);
        const float en  = sE[t * 16 + nlane];
        const int   idx = t * 16 + nlane;
#pragma unroll
        for (int j = 0; j < 8; ++j) {
            // dist = ||e||^2 - 2*(z . e)   (||z||^2 constant per row)
            float d0 = __builtin_fmaf(-2.0f, c0[j], en);
            float d1 = __builtin_fmaf(-2.0f, c1[j], en);
            if (d0 < minv0[j]) { minv0[j] = d0; mini0[j] = idx; }
            if (d1 < minv1[j]) { minv1[j] = d1; mini1[j] = idx; }
        }
    }

    // cross-lane argmin over the 16 columns (stay within 16-lane half)
#pragma unroll
    for (int m = 1; m < 16; m <<= 1) {
#pragma unroll
        for (int j = 0; j < 8; ++j) {
            float ov0 = __shfl_xor(minv0[j], m, 32);
            int   oi0 = __shfl_xor(mini0[j], m, 32);
            if (ov0 < minv0[j] || (ov0 == minv0[j] && oi0 < mini0[j])) {
                minv0[j] = ov0; mini0[j] = oi0;
            }
            float ov1 = __shfl_xor(minv1[j], m, 32);
            int   oi1 = __shfl_xor(mini1[j], m, 32);
            if (ov1 < minv1[j] || (ov1 == minv1[j] && oi1 < mini1[j])) {
                minv1[j] = ov1; mini1[j] = oi1;
            }
        }
    }

    if ((lane & 15) == 0) {
        // lanes 0..15 carry rows M=j, lanes 16..31 carry rows M=j+8
        int half = (lane >> 4) << 3;
        int base0 = tileA0 * 16 + half;
        int base1 = tileA1 * 16 + half;
#pragma unroll
        for (int j = 0; j < 8; ++j) {
            out_idx[base0 + j] = mini0[j];
            out_idx[base1 + j] = mini1[j];
        }
    }
}

// -------- gather z_q, write NCHW + indices, deterministic loss partials
__global__ void gather_kernel(const float* __restrict__ z,
                              const float* __restrict__ table,
                              const int* __restrict__ idx,
                              float* __restrict__ out,      // d_out base
                              float* __restrict__ partials) {
    __shared__ float red[256];
    int n  = blockIdx.x * blockDim.x + threadIdx.x;         // 65536 rows
    int k  = idx[n];
    int b  = n >> 12;
    int hw = n & (HW - 1);
    const float* e = table + k * C_DIM;
    float acc = 0.f;
#pragma unroll
    for (int c = 0; c < C_DIM; ++c) {
        long off = (long)b * CHW + (long)c * HW + hw;       // coalesced in n
        float q  = e[c];
        float dv = z[off] - q;
        acc += dv * dv;
        out[off] = q;                                       // z_q (NCHW)
    }
    out[(long)N_ROWS * C_DIM + n] = (float)k;               // indices

    // deterministic in-block tree reduction
    red[threadIdx.x] = acc;
    __syncthreads();
    for (int s = 128; s > 0; s >>= 1) {
        if (threadIdx.x < s) red[threadIdx.x] += red[threadIdx.x + s];
        __syncthreads();
    }
    if (threadIdx.x == 0) partials[blockIdx.x] = red[0];
}

__global__ void finalize_kernel(const float* __restrict__ partials,
                                float* __restrict__ out) {
    __shared__ float red[256];
    red[threadIdx.x] = partials[threadIdx.x];
    __syncthreads();
    for (int s = 128; s > 0; s >>= 1) {
        if (threadIdx.x < s) red[threadIdx.x] += red[threadIdx.x + s];
        __syncthreads();
    }
    if (threadIdx.x == 0) {
        float mse = red[0] / (float)((long)N_ROWS * C_DIM);
        out[(long)N_ROWS * C_DIM + N_ROWS] = (1.0f + BETA) * mse;
    }
}

extern "C" void kernel_launch(void* const* d_in, const int* in_sizes, int n_in,
                              void* d_out, int out_size, void* d_ws, size_t ws_size,
                              hipStream_t stream) {
    const float* z     = (const float*)d_in[0];   // [16,32,64,64]
    const float* table = (const float*)d_in[1];   // [4096,32]
    float* out = (float*)d_out;

    char* ws = (char*)d_ws;
    float*    partials = (float*)(ws + PARTIAL_OFF);
    float*    enorm    = (float*)(ws + ENORM_OFF);
    _Float16* bpack    = (_Float16*)(ws + BPACK_OFF);
    _Float16* apack    = (_Float16*)(ws + APACK_OFF);
    int*      indices  = (int*)(ws + IDX_OFF);

    enorm_kernel<<<K_CODES / 256, 256, 0, stream>>>(table, enorm);
    packB_kernel<<<256, 256, 0, stream>>>(table, bpack);       // 65536 thr
    packA_kernel<<<4096, 256, 0, stream>>>(z, apack);          // 1M thr

    // 256 blocks * 8 waves * 32 rows = 65536 rows; 272KB dynamic LDS
    vq_argmin_kernel<<<256, 256, 262144 + 16384, stream>>>(
        apack, bpack, enorm, indices);

    gather_kernel<<<N_ROWS / 256, 256, 0, stream>>>(z, table, indices,
                                                    out, partials);
    finalize_kernel<<<1, 256, 0, stream>>>(partials, out);
}
